// PureMambaBlock_40707700031526
// MI455X (gfx1250) — compile-verified, matrix operand order checked by reading, not run on hardware
//
#include <hip/hip_runtime.h>
#include <stdint.h>

// ---------------- problem constants ----------------
#define B_SZ   2
#define LSEQ   2048
#define DMODEL 1024
#define DINNER 2048
#define DSTATE 16
#define DTRANK 64
#define MROWS  (B_SZ * LSEQ)          // 4096 token rows
#define XDBL_C (DTRANK + 2 * DSTATE)  // 96

typedef __attribute__((ext_vector_type(16))) __bf16 v16bf;
typedef __attribute__((ext_vector_type(8)))  float  v8f;

union Frag16 {
  v16bf v;
  uint4 q[2];
};

__device__ __forceinline__ unsigned short f2bf(float f) {
  union { float f; unsigned u; } c;
  c.f = f;
  unsigned u = c.u;
  u += 0x7FFFu + ((u >> 16) & 1u);   // round-to-nearest-even
  return (unsigned short)(u >> 16);
}

__device__ __forceinline__ float siluf(float x) {
  return x / (1.0f + __expf(-x));
}

__device__ __forceinline__ float softplusf(float x) {
  return x > 20.0f ? x : log1pf(__expf(x));
}

// CDNA5 async global->LDS copy (GLOBAL_LOAD_ASYNC_TO_LDS_B128, tracked by ASYNCcnt).
// LDS destination address = low 32 bits of the flat pointer (ISA 10.2: LDS aperture
// addresses are truncated to addr[31:0]).
__device__ __forceinline__ void async_load16_to_lds(const void* gptr, void* lptr) {
  unsigned lds_off = (unsigned)(uintptr_t)lptr;
  unsigned long long ga = (unsigned long long)(uintptr_t)gptr;
  asm volatile("global_load_async_to_lds_b128 %0, %1, off"
               :
               : "v"(lds_off), "v"(ga)
               : "memory");
}

__device__ __forceinline__ void wait_async0() {
  asm volatile("s_wait_asynccnt 0x0" ::: "memory");
}

// ---------------- fp32 -> bf16 convert (strided rows) ----------------
__global__ void cvt_f32_bf16(const float* __restrict__ src,
                             unsigned short* __restrict__ dst,
                             int rows, int ldsrc, int ncols) {
  size_t i = (size_t)blockIdx.x * blockDim.x + threadIdx.x;
  size_t total = (size_t)rows * ncols;
  if (i >= total) return;
  size_t r = i / (size_t)ncols;
  size_t c = i - r * (size_t)ncols;
  dst[i] = f2bf(src[r * (size_t)ldsrc + c]);
}

// ---------------- WMMA GEMM: C[M,N] = A[M,K](bf16) * W[N,K]^T(bf16) ----------------
// Wave tile: 16(M) x JT*16(N); block = 8 waves -> 128 x JT*16 block tile.
// W tile staged to LDS with GLOBAL_LOAD_ASYNC_TO_LDS_B128 (double-buffered; the
// async stage for buffer B overlaps the whole compute of buffer A).
// Protocol per K-step:  s_wait_asynccnt 0  ->  barrier  ->  issue next async stage
// -> load fa (global) + all JT fb frags (LDS) -> JT in-place WMMAs.
// Requires: M % 128 == 0, N % (JT*16) == 0, K % 32 == 0.
// EPI: 0 = plain store, 1 = softplus(v + bias[n]) (dt projection).
template <int EPI, int JT>
__global__ __launch_bounds__(256) void gemm_bf16_wmma(
    const unsigned short* __restrict__ A,
    const unsigned short* __restrict__ W,
    float* __restrict__ C,
    int M, int N, int K,
    const float* __restrict__ bias) {
  __shared__ unsigned short sW[2][JT * 16 * 32];  // [buf][row][k] bf16

  const int wave  = threadIdx.x >> 5;
  const int lane  = threadIdx.x & 31;
  const int tileM = blockIdx.x * 128 + wave * 16;
  const int tileN = blockIdx.y * (JT * 16);
  const int frow  = lane & 15;          // M row (A frag) / N col (B frag) in tile
  const int kgrp  = (lane >> 4) << 3;   // 0 or 8

  const v8f vzero = {0.f, 0.f, 0.f, 0.f, 0.f, 0.f, 0.f, 0.f};
  v8f acc[JT];
#pragma unroll
  for (int j = 0; j < JT; ++j) acc[j] = vzero;

  const unsigned short* aRow = A + (size_t)(tileM + frow) * K + kgrp;

  // Cooperative W staging: 16B chunks; JT*16 rows x 64B per row.
  constexpr int NCHUNK = JT * 16 * 4;
  const int chunk = threadIdx.x;
  const int srow  = chunk >> 2;
  const int ssub  = chunk & 3;
  const unsigned short* wRow = W + (size_t)(tileN + srow) * K + ssub * 8;

  auto stage = [&](int k0, int buf) {
    if (NCHUNK >= 256 || chunk < NCHUNK)
      async_load16_to_lds(wRow + k0, &sW[buf][srow * 32 + ssub * 8]);
  };

  stage(0, 0);
  for (int k0 = 0; k0 < K; k0 += 32) {
    const int cur = (k0 >> 5) & 1;
    wait_async0();      // my async writes into sW[cur] have landed
    __syncthreads();    // everyone's writes visible; prev reads of sW[cur^1] done
    if (k0 + 32 < K) stage(k0 + 32, cur ^ 1);  // overlaps with compute below

    Frag16 fa;
    fa.q[0] = *(const uint4*)(aRow + k0);        // K = kgrp + 0..7
    fa.q[1] = *(const uint4*)(aRow + k0 + 16);   // K = 16 + kgrp + 0..7
    __builtin_prefetch(aRow + k0 + 512, 0, 3);   // global_prefetch_b8 (near)

    Frag16 fb[JT];
#pragma unroll
    for (int j = 0; j < JT; ++j) {
      const unsigned short* bl = &sW[cur][(j * 16 + frow) * 32 + kgrp];
      fb[j].q[0] = *(const uint4*)(bl);
      fb[j].q[1] = *(const uint4*)(bl + 16);
    }
#pragma unroll
    for (int j = 0; j < JT; ++j)
      acc[j] = __builtin_amdgcn_wmma_f32_16x16x32_bf16(
          false, fa.v, false, fb[j].v, (short)0, acc[j], false, false);
  }

  // C/D layout: VGPR r -> M = r + 8*(lane>>4), N = lane&15
  const int rbase = (lane >> 4) << 3;
  const int ncol  = lane & 15;
#pragma unroll
  for (int j = 0; j < JT; ++j) {
    const int n = tileN + j * 16 + ncol;
    const float bv = (EPI == 1) ? bias[n] : 0.f;
#pragma unroll
    for (int r = 0; r < 8; ++r) {
      float v = acc[j][r];
      if (EPI == 1) v = softplusf(v + bv);
      C[(size_t)(tileM + rbase + r) * N + n] = v;
    }
  }
}

// ---------------- depthwise causal conv (width 4) + bias + SiLU ----------------
// xz: [MROWS, 2*DINNER] (cols 0..DINNER-1 = x branch). Writes f32 + bf16 copies.
__global__ __launch_bounds__(256) void conv_silu_kernel(
    const float* __restrict__ xz, const float* __restrict__ cw,
    const float* __restrict__ cb, float* __restrict__ xcf,
    unsigned short* __restrict__ xcb) {
  const int i = blockIdx.x * blockDim.x + threadIdx.x;  // over MROWS*DINNER
  const int row = i / DINNER;
  const int d   = i - row * DINNER;
  const int l   = row & (LSEQ - 1);
  float acc = cb[d];
#pragma unroll
  for (int t = 0; t < 4; ++t) {
    const int ls = l - 3 + t;
    if (ls >= 0)
      acc += xz[(size_t)(row - 3 + t) * (2 * DINNER) + d] * cw[d * 4 + t];
  }
  const float s = siluf(acc);
  xcf[(size_t)row * DINNER + d] = s;
  xcb[(size_t)row * DINNER + d] = f2bf(s);
}

// ---------------- selective scan (serial in L, parallel over b,d) ----------------
// One thread per channel d; h[16] lives in VGPRs; B_t/C_t broadcast via LDS.
// Fuses y = h.C + D*x, gating with silu(z), bf16 output for the out-projection.
__global__ __launch_bounds__(64) void scan_kernel(
    const float* __restrict__ xdbl,   // [MROWS, 96] (dt_low | B | C)
    const float* __restrict__ dtf,    // [MROWS, DINNER] softplus'ed dt
    const float* __restrict__ xcf,    // [MROWS, DINNER] conv+silu output
    const float* __restrict__ xz,     // [MROWS, 2*DINNER] (z = cols DINNER..)
    const float* __restrict__ A_log,  // [DINNER, DSTATE]
    const float* __restrict__ Dp,     // [DINNER]
    unsigned short* __restrict__ ybf) // [MROWS, DINNER]
{
  const int b = blockIdx.x >> 5;                       // 32 blocks per batch
  const int d = ((blockIdx.x & 31) << 6) + threadIdx.x;

  float Arow[DSTATE];
#pragma unroll
  for (int n = 0; n < DSTATE; ++n) Arow[n] = -__expf(A_log[d * DSTATE + n]);
  float h[DSTATE] = {};
  const float Dd = Dp[d];

  __shared__ float sB[DSTATE];
  __shared__ float sC[DSTATE];

  for (int l = 0; l < LSEQ; ++l) {
    const int row = b * LSEQ + l;
    __syncthreads();
    if (threadIdx.x < 32)
      ((threadIdx.x < 16) ? sB[threadIdx.x] : sC[threadIdx.x - 16]) =
          xdbl[(size_t)row * XDBL_C + DTRANK + threadIdx.x];
    __syncthreads();

    const float dt = dtf[(size_t)row * DINNER + d];
    const float xt = xcf[(size_t)row * DINNER + d];
    float y = 0.f;
#pragma unroll
    for (int n = 0; n < DSTATE; ++n) {
      const float dA = __expf(dt * Arow[n]);
      h[n] = dA * h[n] + dt * sB[n] * xt;
      y += h[n] * sC[n];
    }
    y += Dd * xt;
    const float z = xz[(size_t)row * (2 * DINNER) + DINNER + d];
    ybf[(size_t)row * DINNER + d] = f2bf(y * siluf(z));
  }
}

// ---------------- host orchestration ----------------
extern "C" void kernel_launch(void* const* d_in, const int* in_sizes, int n_in,
                              void* d_out, int out_size, void* d_ws, size_t ws_size,
                              hipStream_t stream) {
  const float* x          = (const float*)d_in[0];
  const float* in_proj_w  = (const float*)d_in[1];
  const float* conv_w     = (const float*)d_in[2];
  const float* conv_b     = (const float*)d_in[3];
  const float* x_proj_w   = (const float*)d_in[4];
  const float* dt_proj_w  = (const float*)d_in[5];
  const float* dt_proj_b  = (const float*)d_in[6];
  const float* A_log      = (const float*)d_in[7];
  const float* D_param    = (const float*)d_in[8];
  const float* out_proj_w = (const float*)d_in[9];
  float* out = (float*)d_out;

  char* ws = (char*)d_ws;
  size_t off = 0;
  auto alloc = [&](size_t bytes) {
    void* p = ws + off;
    off = (off + bytes + 255) & ~(size_t)255;
    return p;
  };

  unsigned short* xbf        = (unsigned short*)alloc((size_t)MROWS * DMODEL * 2);
  unsigned short* w_in_bf    = (unsigned short*)alloc((size_t)2 * DINNER * DMODEL * 2);
  unsigned short* w_xproj_bf = (unsigned short*)alloc((size_t)XDBL_C * DINNER * 2);
  unsigned short* w_dt_bf    = (unsigned short*)alloc((size_t)DINNER * DTRANK * 2);
  unsigned short* w_out_bf   = (unsigned short*)alloc((size_t)DMODEL * DINNER * 2);
  float*          xzf        = (float*)alloc((size_t)MROWS * 2 * DINNER * 4);
  float*          xcf        = (float*)alloc((size_t)MROWS * DINNER * 4);
  unsigned short* xcb        = (unsigned short*)alloc((size_t)MROWS * DINNER * 2);
  float*          xdbl       = (float*)alloc((size_t)MROWS * XDBL_C * 4);
  unsigned short* dtbf       = (unsigned short*)alloc((size_t)MROWS * DTRANK * 2);
  float*          dtf        = (float*)alloc((size_t)MROWS * DINNER * 4);
  unsigned short* ybf        = (unsigned short*)alloc((size_t)MROWS * DINNER * 2);

  auto cvt = [&](const float* s, unsigned short* dst, int rows, int ld, int nc) {
    size_t tot = (size_t)rows * nc;
    cvt_f32_bf16<<<(unsigned)((tot + 255) / 256), 256, 0, stream>>>(s, dst, rows, ld, nc);
  };

  // 1) bf16 conversions for WMMA operands
  cvt(x, xbf, MROWS, DMODEL, DMODEL);
  cvt(in_proj_w, w_in_bf, 2 * DINNER, DMODEL, DMODEL);
  cvt(x_proj_w, w_xproj_bf, XDBL_C, DINNER, DINNER);
  cvt(dt_proj_w, w_dt_bf, DINNER, DTRANK, DTRANK);
  cvt(out_proj_w, w_out_bf, DMODEL, DINNER, DINNER);

  // 2) in_proj: xz[4096,4096] = xbf[4096,1024] * w_in^T   (JT=4 -> N tile 64)
  gemm_bf16_wmma<0, 4><<<dim3(MROWS / 128, (2 * DINNER) / 64), 256, 0, stream>>>(
      xbf, w_in_bf, xzf, MROWS, 2 * DINNER, DMODEL, nullptr);

  // 3) depthwise causal conv + SiLU
  conv_silu_kernel<<<(MROWS * DINNER) / 256, 256, 0, stream>>>(
      xzf, conv_w, conv_b, xcf, xcb);

  // 4) x_proj: xdbl[4096,96] = xcb[4096,2048] * w_xproj^T (JT=2 -> N tile 32, 96=3*32)
  gemm_bf16_wmma<0, 2><<<dim3(MROWS / 128, XDBL_C / 32), 256, 0, stream>>>(
      xcb, w_xproj_bf, xdbl, MROWS, XDBL_C, DINNER, nullptr);

  // 5) dt_low slice -> bf16
  cvt(xdbl, dtbf, MROWS, XDBL_C, DTRANK);

  // 6) dt_proj + bias + softplus: dtf[4096,2048]
  gemm_bf16_wmma<1, 4><<<dim3(MROWS / 128, DINNER / 64), 256, 0, stream>>>(
      dtbf, w_dt_bf, dtf, MROWS, DINNER, DTRANK, dt_proj_b);

  // 7) selective scan + D*x + silu(z) gating -> ybf (bf16)
  scan_kernel<<<B_SZ * (DINNER / 64), 64, 0, stream>>>(
      xdbl, dtf, xcf, xzf, A_log, D_param, ybf);

  // 8) out_proj: out[4096,1024] = ybf[4096,2048] * w_out^T
  gemm_bf16_wmma<0, 4><<<dim3(MROWS / 128, DMODEL / 64), 256, 0, stream>>>(
      ybf, w_out_bf, out, MROWS, DMODEL, DINNER, nullptr);
}
